// AGCN_GNN_704374636678
// MI455X (gfx1250) — matrix operand order repaired and checked by reading, not compiled, not run on hardware
//
#include <hip/hip_runtime.h>
#include <hip/hip_bf16.h>

// ---------------------------------------------------------------------------
// AGCN forward for MI455X (gfx1250, wave32).
//  * Dense GEMMs: v_wmma_f32_16x16x32_bf16, A pre-converted to zero-padded
//    bf16 (no conversion VALU in the hot loop), B tile staged in LDS via
//    global_load_async_to_lds_b128 + s_wait_asynccnt (CDNA5 async path).
//  * SpMM: COO edge-parallel f32 atomics (memory/atomic bound, HBM @23.3TB/s).
//  * Gating MLPs / softmax: wave-per-node with __shfl_xor reductions.
// ---------------------------------------------------------------------------

typedef __bf16 bf16_t;
typedef __attribute__((ext_vector_type(16))) __bf16 v16bf;
typedef __attribute__((ext_vector_type(8)))  __bf16 v8bf;
typedef __attribute__((ext_vector_type(8)))  float  v8f;

#define NNODES 50000
#define NEDGES 1600000
#define BLK_M  128              // 8 waves x 16 rows
#define BLK_N  64               // 4 x 16-col WMMA tiles per wave
#define MPAD   50048            // NNODES rounded up to BLK_M (391 * 128)

__device__ __forceinline__ bf16_t f2bf(float f) {
  union { float f; unsigned u; } v; v.f = f;
  unsigned r = v.u + 0x7FFFu + ((v.u >> 16) & 1u);   // round-to-nearest-even
  unsigned short h = (unsigned short)(r >> 16);
  bf16_t out;
  __builtin_memcpy(&out, &h, sizeof(out));
  return out;
}

__device__ __forceinline__ float wave_sum32(float v) {
#pragma unroll
  for (int off = 16; off > 0; off >>= 1) v += __shfl_xor(v, off, 32);
  return v;
}

// Low 32 bits of a flat LDS pointer == wave-relative LDS byte address
// (CDNA5 aperture: high 32 bits select LDS, low 32 bits are the LDS address).
__device__ __forceinline__ unsigned lds_addr_of(const void* p) {
  return (unsigned)(uintptr_t)p;
}

// ---------------------------------------------------------------------------
// Weight convert: W (K x N, f32, row-major) -> Wt (Npad x Kpad, bf16,
// transposed, zero-padded) so the GEMM loads B unguarded.
// ---------------------------------------------------------------------------
__global__ __launch_bounds__(256) void k_convert_wt(
    const float* __restrict__ W, bf16_t* __restrict__ Wt,
    int K, int Ncols, int Kpad, int Npad) {
  long long total = (long long)Npad * Kpad;
  long long stride = (long long)gridDim.x * blockDim.x;
  for (long long idx = (long long)blockIdx.x * blockDim.x + threadIdx.x;
       idx < total; idx += stride) {
    int n = (int)(idx / Kpad);
    int k = (int)(idx % Kpad);
    float v = (n < Ncols && k < K) ? W[(size_t)k * Ncols + n] : 0.0f;
    Wt[idx] = f2bf(v);
  }
}

// A convert: A (M x K f32) -> Abf (Mpad x Kpad bf16, zero-padded)
__global__ __launch_bounds__(256) void k_convert_a(
    const float* __restrict__ A, bf16_t* __restrict__ Abf,
    int M, int K, int Kpad, int Mpad) {
  long long total = (long long)Mpad * Kpad;
  long long stride = (long long)gridDim.x * blockDim.x;
  for (long long idx = (long long)blockIdx.x * blockDim.x + threadIdx.x;
       idx < total; idx += stride) {
    int m = (int)(idx / Kpad);
    int k = (int)(idx % Kpad);
    float v = (m < M && k < K) ? A[(size_t)m * K + k] : 0.0f;
    Abf[idx] = f2bf(v);
  }
}

// ---------------------------------------------------------------------------
// WMMA GEMM: C[M x N] = Abf[Mpad x Kpad] * Wt^T   (both bf16, zero-padded)
// Block = 256 threads (8 waves). Block tile 128(M) x 64(N); each wave owns a
// 16x64 strip = 4 WMMA tiles sharing one A fragment per 32-K step.
// B tile (64n x 32k bf16 = 4KB) staged in LDS via async global->LDS copy.
// C/D layout per CDNA5 ISA 7.12.2; A/B use contiguous k = kh*16 + s (wave32).
// ---------------------------------------------------------------------------
__global__ __launch_bounds__(256) void k_wmma_gemm(
    const bf16_t* __restrict__ A, const bf16_t* __restrict__ Bt,
    float* __restrict__ C, int M, int N, int Kpad, int ldc) {
  __shared__ __attribute__((aligned(64))) bf16_t tileB[BLK_N * 32];  // 4 KB

  const int lane = threadIdx.x & 31;
  const int wave = threadIdx.x >> 5;
  const int l16  = lane & 15;
  const int kh   = lane >> 4;                 // K-half selector (contiguous k = kh*16 + s)
  const int m_base = blockIdx.x * BLK_M + wave * 16;
  const int n_base = blockIdx.y * BLK_N;

  // Async-copy assignment: thread t moves one 16B chunk of the B tile.
  const int br = threadIdx.x >> 2;            // 0..63 : n-row within tile
  const int bc = threadIdx.x & 3;             // 0..3  : 16B chunk within row
  const bf16_t* bsrc = Bt + (size_t)(n_base + br) * Kpad + bc * 8;
  const unsigned lds_dst = lds_addr_of(&tileB[br * 32 + bc * 8]);

  const bf16_t* Arow = A + (size_t)(m_base + l16) * Kpad;

  v8f acc[4];
#pragma unroll
  for (int j = 0; j < 4; ++j) acc[j] = (v8f){};

  for (int k0 = 0; k0 < Kpad; k0 += 32) {
    __syncthreads();                           // previous reads of tileB done
    asm volatile("global_load_async_to_lds_b128 %0, %1, off"
                 :: "v"(lds_dst),
                    "v"((unsigned long long)(uintptr_t)(bsrc + k0))
                 : "memory");
    asm volatile("s_wait_asynccnt 0x0" ::: "memory");
    __syncthreads();                           // whole B tile visible

    // A fragment: lane l16 = row, elements 0..15 -> k = kh*16 + j
    // (two contiguous 16B loads)
    const bf16_t* pa = Arow + k0 + kh * 16;
    __builtin_prefetch(pa + 64, 0, 0);         // global_prefetch_b8
    v8bf alo = *(const v8bf*)(pa);
    v8bf ahi = *(const v8bf*)(pa + 8);
    v16bf a = __builtin_shufflevector(alo, ahi,
        0, 1, 2, 3, 4, 5, 6, 7, 8, 9, 10, 11, 12, 13, 14, 15);

    // B fragments from LDS: lane column n = j*16 + l16, k = kh*16 .. +15
    const bf16_t* bp = &tileB[l16 * 32 + kh * 16];
#pragma unroll
    for (int j = 0; j < 4; ++j) {
      v16bf b = *(const v16bf*)(bp + j * 16 * 32);
      acc[j] = __builtin_amdgcn_wmma_f32_16x16x32_bf16(
          false, a, false, b, (short)0, acc[j], false, false);
    }
  }

  // C/D layout: lane<16 -> rows m_base+0..7 ; lane>=16 -> rows m_base+8..15
  if (m_base + 16 <= M) {                      // wave-uniform, after barriers
    const int rbase = m_base + (kh << 3);
#pragma unroll
    for (int j = 0; j < 4; ++j) {
      int n = n_base + j * 16 + l16;
      if (n < N) {
#pragma unroll
        for (int i = 0; i < 8; ++i)
          C[(size_t)(rbase + i) * ldc + n] = acc[j][i];
      }
    }
  }
}

// ---------------------------------------------------------------------------
// SpMM: Y[r,:] += val * X[c,:]  (COO, edge x feature parallel, f32 atomics)
// ---------------------------------------------------------------------------
__global__ __launch_bounds__(256) void k_spmm(
    const int* __restrict__ rows, const int* __restrict__ cols,
    const float* __restrict__ vals, const float* __restrict__ X,
    float* __restrict__ Y, long long E, int d) {
  long long total = E * d;
  long long stride = (long long)gridDim.x * blockDim.x;
  for (long long idx = (long long)blockIdx.x * blockDim.x + threadIdx.x;
       idx < total; idx += stride) {
    long long e = idx / d;
    int f = (int)(idx % d);
    int r = rows[e], c = cols[e];
    atomicAdd(&Y[(size_t)r * d + f], vals[e] * X[(size_t)c * d + f]);
  }
}

__global__ __launch_bounds__(256) void k_relu(float* __restrict__ p, long long n) {
  long long stride = (long long)gridDim.x * blockDim.x;
  for (long long i = (long long)blockIdx.x * blockDim.x + threadIdx.x; i < n; i += stride)
    p[i] = fmaxf(p[i], 0.0f);
}

// out = m[:,0]*zf + m[:,1]*h
__global__ __launch_bounds__(256) void k_mix(
    const float* __restrict__ m, const float* __restrict__ zf,
    const float* __restrict__ h, float* __restrict__ out,
    long long total, int d) {
  long long stride = (long long)gridDim.x * blockDim.x;
  for (long long i = (long long)blockIdx.x * blockDim.x + threadIdx.x; i < total; i += stride) {
    long long node = i / d;
    out[i] = m[node * 2 + 0] * zf[i] + m[node * 2 + 1] * h[i];
  }
}

// gate: m = L2norm(softmax(leaky_relu(cat(h,z) @ Wm + bm)))  — wave per node
__global__ __launch_bounds__(256) void k_gate(
    const float* __restrict__ h, const float* __restrict__ zf,
    const float* __restrict__ Wm, const float* __restrict__ bm,
    float* __restrict__ m, int n, int d) {
  int node = blockIdx.x * 8 + (threadIdx.x >> 5);
  int lane = threadIdx.x & 31;
  if (node >= n) return;
  float a0 = 0.0f, a1 = 0.0f;
  int twoD = 2 * d;
  for (int i = lane; i < twoD; i += 32) {
    float v = (i < d) ? h[(size_t)node * d + i] : zf[(size_t)node * d + (i - d)];
    a0 += v * Wm[i * 2 + 0];
    a1 += v * Wm[i * 2 + 1];
  }
  a0 = wave_sum32(a0);
  a1 = wave_sum32(a1);
  if (lane == 0) {
    float l0 = a0 + bm[0], l1 = a1 + bm[1];
    l0 = (l0 > 0.0f) ? l0 : 0.01f * l0;
    l1 = (l1 > 0.0f) ? l1 : 0.01f * l1;
    float mx = fmaxf(l0, l1);
    float e0 = __expf(l0 - mx), e1 = __expf(l1 - mx);
    float s = e0 + e1; e0 /= s; e1 /= s;
    float nrm = fmaxf(sqrtf(e0 * e0 + e1 * e1), 1e-12f);
    m[node * 2 + 0] = e0 / nrm;
    m[node * 2 + 1] = e1 / nrm;
  }
}

// virtual concatenation cat = [z1(500) z2(500) z3(2000) z4(10) z(10)]
__device__ __forceinline__ float cat_val(int k, int node,
    const float* z1, const float* z2, const float* z3,
    const float* z4, const float* zin) {
  if (k < 500)  return z1[(size_t)node * 500 + k];
  if (k < 1000) return z2[(size_t)node * 500 + (k - 500)];
  if (k < 3000) return z3[(size_t)node * 2000 + (k - 1000)];
  if (k < 3010) return z4[node * 10 + (k - 3000)];
  return zin[node * 10 + (k - 3010)];
}

__device__ __forceinline__ int cat_seg(int k) {
  if (k < 500)  return 0;
  if (k < 1000) return 1;
  if (k < 3000) return 2;
  if (k < 3010) return 3;
  return 4;
}

// u = L2norm(softmax(leaky_relu(cat @ Wml + bml)))  — wave per node
__global__ __launch_bounds__(256) void k_ugate(
    const float* __restrict__ z1, const float* __restrict__ z2,
    const float* __restrict__ z3, const float* __restrict__ z4,
    const float* __restrict__ zin, const float* __restrict__ Wml,
    const float* __restrict__ bml, float* __restrict__ u, int n) {
  int node = blockIdx.x * 8 + (threadIdx.x >> 5);
  int lane = threadIdx.x & 31;
  if (node >= n) return;
  float acc[5] = {0, 0, 0, 0, 0};
  for (int k = lane; k < 3020; k += 32) {
    float a = cat_val(k, node, z1, z2, z3, z4, zin);
#pragma unroll
    for (int j = 0; j < 5; ++j) acc[j] += a * Wml[k * 5 + j];
  }
#pragma unroll
  for (int j = 0; j < 5; ++j) acc[j] = wave_sum32(acc[j]);
  if (lane == 0) {
    float l[5], mx = -3.4e38f;
#pragma unroll
    for (int j = 0; j < 5; ++j) {
      float v = acc[j] + bml[j];
      l[j] = (v > 0.0f) ? v : 0.01f * v;
      mx = fmaxf(mx, l[j]);
    }
    float s = 0.0f;
#pragma unroll
    for (int j = 0; j < 5; ++j) { l[j] = __expf(l[j] - mx); s += l[j]; }
    float nrm = 0.0f;
#pragma unroll
    for (int j = 0; j < 5; ++j) { l[j] /= s; nrm += l[j] * l[j]; }
    nrm = fmaxf(sqrtf(nrm), 1e-12f);
#pragma unroll
    for (int j = 0; j < 5; ++j) u[node * 5 + j] = l[j] / nrm;
  }
}

// head: out[node,c] = sum_k u_seg(k)*cat(k)*W5[k,c]  (N=10, K=3020) — wave/node
__global__ __launch_bounds__(256) void k_fused_logits(
    const float* __restrict__ z1, const float* __restrict__ z2,
    const float* __restrict__ z3, const float* __restrict__ z4,
    const float* __restrict__ zin, const float* __restrict__ u,
    const float* __restrict__ W5, float* __restrict__ out, int n) {
  int node = blockIdx.x * 8 + (threadIdx.x >> 5);
  int lane = threadIdx.x & 31;
  if (node >= n) return;
  float u0 = u[node * 5 + 0], u1 = u[node * 5 + 1], u2 = u[node * 5 + 2];
  float u3 = u[node * 5 + 3], u4 = u[node * 5 + 4];
  float acc[10] = {0, 0, 0, 0, 0, 0, 0, 0, 0, 0};
  for (int k = lane; k < 3020; k += 32) {
    int seg = cat_seg(k);
    float us = (seg == 0) ? u0 : (seg == 1) ? u1 : (seg == 2) ? u2 : (seg == 3) ? u3 : u4;
    float a = us * cat_val(k, node, z1, z2, z3, z4, zin);
#pragma unroll
    for (int c = 0; c < 10; ++c) acc[c] += a * W5[k * 10 + c];
  }
#pragma unroll
  for (int c = 0; c < 10; ++c) acc[c] = wave_sum32(acc[c]);
  if (lane == 0) {
#pragma unroll
    for (int c = 0; c < 10; ++c) out[(size_t)node * 10 + c] = acc[c];
  }
}

__global__ __launch_bounds__(256) void k_softmax10(
    const float* __restrict__ L, float* __restrict__ out, int n) {
  int i = blockIdx.x * 256 + threadIdx.x;
  if (i >= n) return;
  float v[10], mx = -3.4e38f;
#pragma unroll
  for (int c = 0; c < 10; ++c) { v[c] = L[(size_t)i * 10 + c]; mx = fmaxf(mx, v[c]); }
  float s = 0.0f;
#pragma unroll
  for (int c = 0; c < 10; ++c) { v[c] = __expf(v[c] - mx); s += v[c]; }
#pragma unroll
  for (int c = 0; c < 10; ++c) out[(size_t)i * 10 + c] = v[c] / s;
}

// ---------------------------------------------------------------------------
extern "C" void kernel_launch(void* const* d_in, const int* in_sizes, int n_in,
                              void* d_out, int out_size, void* d_ws, size_t ws_size,
                              hipStream_t stream) {
  (void)in_sizes; (void)n_in; (void)out_size; (void)ws_size;
  const float* x    = (const float*)d_in[0];
  const int*   rows = (const int*)d_in[1];
  const int*   cols = (const int*)d_in[2];
  const float* vals = (const float*)d_in[3];
  const float* h1   = (const float*)d_in[4];
  const float* h2   = (const float*)d_in[5];
  const float* h3   = (const float*)d_in[6];
  const float* zin  = (const float*)d_in[7];
  const float* W1   = (const float*)d_in[8];
  const float* W2   = (const float*)d_in[9];
  const float* W3   = (const float*)d_in[10];
  const float* W4   = (const float*)d_in[11];
  const float* W5   = (const float*)d_in[12];
  const float* Wm1  = (const float*)d_in[13];
  const float* bm1  = (const float*)d_in[14];
  const float* Wm2  = (const float*)d_in[15];
  const float* bm2  = (const float*)d_in[16];
  const float* Wm3  = (const float*)d_in[17];
  const float* bm3  = (const float*)d_in[18];
  const float* Wml  = (const float*)d_in[19];
  const float* bml  = (const float*)d_in[20];

  // Workspace carve-up (~1.62 GB). All bf16 regions 64B-aligned.
  float* ws = (float*)d_ws;
  float* z1   = ws; ws += (long long)NNODES * 500;    // 100 MB
  float* z2   = ws; ws += (long long)NNODES * 500;    // 100 MB
  float* z3   = ws; ws += (long long)NNODES * 2000;   // 400 MB
  float* z4   = ws; ws += (long long)NNODES * 16;
  float* P    = ws; ws += (long long)NNODES * 2000;   // 400 MB (GEMM out)
  float* MIX  = ws; ws += (long long)NNODES * 2000;   // 400 MB (gated input)
  float* m1   = ws; ws += (long long)NNODES * 2;
  float* m2   = ws; ws += (long long)NNODES * 2;
  float* m3   = ws; ws += (long long)NNODES * 2;
  float* u    = ws; ws += (long long)NNODES * 5;
  float* Lbuf = ws; ws += (long long)NNODES * 10 + 16;
  ws = (float*)(((uintptr_t)ws + 63) & ~(uintptr_t)63);
  bf16_t* Wt  = (bf16_t*)ws; ws += 2 * 1024 * 1024;   // 8 MB bf16 (Npad*Kpad max ~1.05M)
  bf16_t* Abf = (bf16_t*)ws;                          // Mpad x Kpad bf16, <= 202 MB

  const int EWBLK = 262144;     // grid-stride cap for elementwise kernels
  const int NWAVEBLK = (NNODES + 7) / 8;

  auto gemm = [&](const float* Af32, const float* W, int K, int N, float* C) {
    int Kpad = (K + 31) & ~31;
    int Npad = (N + 63) & ~63;
    long long tw = (long long)Npad * Kpad;
    int cb = (int)((tw + 255) / 256);
    k_convert_wt<<<cb, 256, 0, stream>>>(W, Wt, K, N, Kpad, Npad);
    long long ta = (long long)MPAD * Kpad;
    long long ab = (ta + 255) / 256;
    int nab = (int)(ab < EWBLK ? ab : EWBLK);
    k_convert_a<<<nab, 256, 0, stream>>>(Af32, Abf, NNODES, K, Kpad, MPAD);
    dim3 g(MPAD / BLK_M, Npad / BLK_N);
    k_wmma_gemm<<<g, 256, 0, stream>>>(Abf, Wt, C, NNODES, N, Kpad, N);
  };

  auto spmm = [&](const float* X, float* Y, int d, bool relu) {
    hipMemsetAsync(Y, 0, (size_t)NNODES * d * sizeof(float), stream);
    long long total = (long long)NEDGES * d;
    long long b = (total + 255) / 256;
    int nb = (int)(b < 524288 ? b : 524288);
    k_spmm<<<nb, 256, 0, stream>>>(rows, cols, vals, X, Y, (long long)NEDGES, d);
    if (relu) {
      long long ne = (long long)NNODES * d;
      long long rb = (ne + 255) / 256;
      int nrb = (int)(rb < EWBLK ? rb : EWBLK);
      k_relu<<<nrb, 256, 0, stream>>>(Y, ne);
    }
  };

  auto mix = [&](const float* mg, const float* zf, const float* h, float* out, int d) {
    long long total = (long long)NNODES * d;
    long long b = (total + 255) / 256;
    int nb = (int)(b < EWBLK ? b : EWBLK);
    k_mix<<<nb, 256, 0, stream>>>(mg, zf, h, out, total, d);
  };

  // Layer 1: z1 = relu(spmm(x @ W1))
  gemm(x, W1, 2000, 500, P);
  spmm(P, z1, 500, true);
  k_gate<<<NWAVEBLK, 256, 0, stream>>>(h1, z1, Wm1, bm1, m1, NNODES, 500);
  mix(m1, z1, h1, MIX, 500);

  // Layer 2
  gemm(MIX, W2, 500, 500, P);
  spmm(P, z2, 500, true);
  k_gate<<<NWAVEBLK, 256, 0, stream>>>(h2, z2, Wm2, bm2, m2, NNODES, 500);
  mix(m2, z2, h2, MIX, 500);

  // Layer 3
  gemm(MIX, W3, 500, 2000, P);
  spmm(P, z3, 2000, true);
  k_gate<<<NWAVEBLK, 256, 0, stream>>>(h3, z3, Wm3, bm3, m3, NNODES, 2000);
  mix(m3, z3, h3, MIX, 2000);

  // Layer 4
  gemm(MIX, W4, 2000, 10, P);
  spmm(P, z4, 10, true);

  // Attention over concatenated embeddings + cluster head
  k_ugate<<<NWAVEBLK, 256, 0, stream>>>(z1, z2, z3, z4, zin, Wml, bml, u, NNODES);
  k_fused_logits<<<NWAVEBLK, 256, 0, stream>>>(z1, z2, z3, z4, zin, u, W5, P, NNODES);
  spmm(P, Lbuf, 10, false);
  k_softmax10<<<(NNODES + 255) / 256, 256, 0, stream>>>(Lbuf, (float*)d_out, NNODES);
}
